// GroupQueryAttention_37580963840420
// MI455X (gfx1250) — matrix operand order repaired and checked by reading, not compiled, not run on hardware
//
#include <hip/hip_runtime.h>
#include <hip/hip_bf16.h>

typedef __bf16 bf16;
typedef __attribute__((ext_vector_type(16))) __bf16 v16bf;
typedef __attribute__((ext_vector_type(8)))  __bf16 v8bf;
typedef __attribute__((ext_vector_type(8)))  float  v8f;
typedef __attribute__((ext_vector_type(4)))  float  f32x4;

constexpr int B_   = 2;
constexpr int S_   = 2048;
constexpr int H_   = 1024;
constexpr int NH_  = 16;
constexpr int G_   = 4;
constexpr int HD_  = 64;
constexpr int HPG_ = 4;
constexpr int KVW_ = G_ * HD_;   // 256

__device__ __forceinline__ bf16 tobf(float x) { return (bf16)x; }

__device__ __forceinline__ v16bf cat8(v8bf lo, v8bf hi) {
  return __builtin_shufflevector(lo, hi, 0,1,2,3,4,5,6,7,8,9,10,11,12,13,14,15);
}

__device__ __forceinline__ v8f zero8() {
  v8f r;
#pragma unroll
  for (int i = 0; i < 8; ++i) r[i] = 0.0f;
  return r;
}

__device__ __forceinline__ v8f wmma_bf16(v16bf a, v16bf b, v8f c) {
  return __builtin_amdgcn_wmma_f32_16x16x32_bf16(false, a, false, b, (short)0, c,
                                                 false, false);
}

// Async copy 16 bytes global -> LDS (ASYNCcnt-tracked, CDNA5).
__device__ __forceinline__ void async_copy_b128(const void* g, void* lds) {
  const unsigned            l  = (unsigned)(unsigned long long)lds;   // LDS offset
  const unsigned long long  ga = (unsigned long long)g;
  asm volatile("global_load_async_to_lds_b128 %0, %1, off"
               :: "v"(l), "v"(ga) : "memory");
}
__device__ __forceinline__ void wait_asynccnt0() {
  asm volatile("s_wait_asynccnt 0" ::: "memory");
}

// ---------------------------------------------------------------------------
// C[M,N] = A[M,K] @ W[K,N] + bias[N]
// AT = float  : A converted fp32->bf16 while staging (batched loads)
// AT = bf16   : A staged with global_load_async_to_lds_b128 (pure copy)
// BF16OUT     : store C as bf16 (intermediates) else fp32
// BM=BN=128, BK=32, 256 thr = 8 waves (2x4); wave tile 64x32; double-buffered.
// ---------------------------------------------------------------------------
template <typename AT, bool BF16OUT>
__global__ __launch_bounds__(256) void gemm_bias_kernel(
    const AT* __restrict__ A, const float* __restrict__ W,
    const float* __restrict__ bias, void* __restrict__ Cout,
    int M, int N, int K)
{
  constexpr int BM = 128, BN = 128, BK = 32;
  constexpr bool A_F32 = (sizeof(AT) == 4);
  __shared__ __align__(16) bf16 As[2][BM][BK];    // [m][k]
  __shared__ __align__(16) bf16 WsT[2][BN][BK];   // W^T tile [n][k]

  const int t    = threadIdx.x;
  const int wave = t >> 5;
  const int lane = t & 31;
  const int hf   = lane >> 4;
  const int lr   = lane & 15;
  const int wm   = (wave >> 2) * 64;
  const int wn   = (wave & 3) * 32;
  const long m0  = (long)blockIdx.x * BM;
  const long n0  = (long)blockIdx.y * BN;

  v8f acc[4][2];
#pragma unroll
  for (int i = 0; i < 4; ++i)
#pragma unroll
    for (int j = 0; j < 2; ++j) acc[i][j] = zero8();

  auto stage = [&](int buf, int k0) {
    if constexpr (A_F32) {
      const int arow = t >> 3, acol = (t & 7) * 4;
      f32x4 av[4];
#pragma unroll
      for (int p = 0; p < 4; ++p)
        av[p] = *(const f32x4*)(A + (m0 + arow + p * 32) * K + k0 + acol);
#pragma unroll
      for (int p = 0; p < 4; ++p) {
        bf16* d = &As[buf][arow + p * 32][acol];
        d[0] = tobf(av[p].x); d[1] = tobf(av[p].y);
        d[2] = tobf(av[p].z); d[3] = tobf(av[p].w);
      }
    } else {
      const int arow = t >> 2, acol = (t & 3) * 8;
#pragma unroll
      for (int p = 0; p < 2; ++p)
        async_copy_b128(A + (m0 + arow + p * 64) * K + k0 + acol,
                        &As[buf][arow + p * 64][acol]);
    }
    {
      const int wrow = t >> 5, wcol = (t & 31) * 4;
      f32x4 wv[4];
#pragma unroll
      for (int p = 0; p < 4; ++p)
        wv[p] = *(const f32x4*)(W + (long)(k0 + wrow + p * 8) * N + n0 + wcol);
#pragma unroll
      for (int p = 0; p < 4; ++p) {
        const int kk = wrow + p * 8;
        WsT[buf][wcol + 0][kk] = tobf(wv[p].x);
        WsT[buf][wcol + 1][kk] = tobf(wv[p].y);
        WsT[buf][wcol + 2][kk] = tobf(wv[p].z);
        WsT[buf][wcol + 3][kk] = tobf(wv[p].w);
      }
    }
  };

  const int nk = K / BK;
  stage(0, 0);
  for (int i = 0; i < nk; ++i) {
    wait_asynccnt0();          // publish async writes of stage(i)
    __syncthreads();
    if (i + 1 < nk) stage((i + 1) & 1, (i + 1) * BK);  // prefetch next tile
    const int buf = i & 1;

    v16bf af[4];
#pragma unroll
    for (int ii = 0; ii < 4; ++ii) {
      const bf16* rp = &As[buf][wm + ii * 16 + lr][0];
      af[ii] = cat8(*(const v8bf*)(rp + hf * 8),
                    *(const v8bf*)(rp + 16 + hf * 8));
    }
    v16bf wb[2];
#pragma unroll
    for (int j = 0; j < 2; ++j) {
      const bf16* cp = &WsT[buf][wn + j * 16 + lr][0];
      wb[j] = cat8(*(const v8bf*)(cp + hf * 8),
                   *(const v8bf*)(cp + 16 + hf * 8));
    }
#pragma unroll
    for (int ii = 0; ii < 4; ++ii)
#pragma unroll
      for (int j = 0; j < 2; ++j)
        acc[ii][j] = wmma_bf16(af[ii], wb[j], acc[ii][j]);
  }

#pragma unroll
  for (int i = 0; i < 4; ++i) {
    const long gm = m0 + wm + i * 16 + hf * 8;
#pragma unroll
    for (int j = 0; j < 2; ++j) {
      const long gn = n0 + wn + j * 16 + lr;
      const float bv = bias[gn];
      if constexpr (BF16OUT) {
        bf16* cp = (bf16*)Cout + gm * N + gn;
#pragma unroll
        for (int v = 0; v < 8; ++v) cp[(long)v * N] = tobf(acc[i][j][v] + bv);
      } else {
        float* cp = (float*)Cout + gm * N + gn;
#pragma unroll
        for (int v = 0; v < 8; ++v) cp[(long)v * N] = acc[i][j][v] + bv;
      }
    }
  }
}

// ---------------------------------------------------------------------------
// Flash attention (causal), bf16 in/out. 256 thr = 8 waves; wave w owns query
// rows [q0+16w, q0+16w+16). KB=64 keys per block, double-buffered staging:
// K via async global->LDS DMA, V transposed via VGPRs. Row max via
// interleaved width-16 shuffles; row sum via WMMA against an all-ones B.
// ---------------------------------------------------------------------------
__global__ __launch_bounds__(256) void gqa_flash_kernel(
    const bf16* __restrict__ Q, const bf16* __restrict__ Kp,
    const bf16* __restrict__ Vp, bf16* __restrict__ O)
{
  constexpr int KB = 64;
  __shared__ __align__(16) bf16 Ks[2][KB][HD_];    // [key][d]
  __shared__ __align__(16) bf16 VsT[2][HD_][KB];   // [d][key]
  __shared__ __align__(16) bf16 Ps[8][16][KB];     // per-wave P scratch [m][key]

  const int t    = threadIdx.x;
  const int wave = t >> 5;
  const int lane = t & 31;
  const int hf   = lane >> 4;
  const int lr   = lane & 15;
  const int q0   = blockIdx.x * 128;
  const int h    = blockIdx.y;
  const int b    = blockIdx.z;
  const int g    = h / HPG_;
  const int qt   = q0 + wave * 16;
  const float scale = 0.125f;   // 1/sqrt(64)

  const bf16* qb  = Q  + (long)b * S_ * H_   + (long)h * HD_;
  const bf16* kbp = Kp + (long)b * S_ * KVW_ + (long)g * HD_;
  const bf16* vbp = Vp + (long)b * S_ * KVW_ + (long)g * HD_;
  bf16*       ob  = O  + (long)b * S_ * H_   + (long)h * HD_;

  // Q tile -> 2 A-fragments
  v16bf qa[2];
  {
    const bf16* qr = qb + (long)(qt + lr) * H_;
#pragma unroll
    for (int kf = 0; kf < 2; ++kf)
      qa[kf] = cat8(*(const v8bf*)(qr + kf * 32 + hf * 8),
                    *(const v8bf*)(qr + kf * 32 + 16 + hf * 8));
  }

  // all-ones B fragment for WMMA row sums
  v16bf ones;
#pragma unroll
  for (int i = 0; i < 16; ++i) ones[i] = (bf16)1.0f;

  v8f o[4];
#pragma unroll
  for (int i = 0; i < 4; ++i) o[i] = zero8();
  float rmax[8], rsum[8];
#pragma unroll
  for (int v = 0; v < 8; ++v) { rmax[v] = -1e30f; rsum[v] = 0.0f; }

  auto stageKV = [&](int buf, int kbeg) {
    const int r  = t >> 2;         // key 0..63
    const int c0 = (t & 3) * 16;   // d 0,16,32,48
    const bf16* kg = kbp + (long)(kbeg + r) * KVW_ + c0;
    async_copy_b128(kg,     &Ks[buf][r][c0]);
    async_copy_b128(kg + 8, &Ks[buf][r][c0 + 8]);
    const bf16* vg = vbp + (long)(kbeg + r) * KVW_ + c0;
    v8bf vv0 = *(const v8bf*)vg;
    v8bf vv1 = *(const v8bf*)(vg + 8);
#pragma unroll
    for (int j = 0; j < 8; ++j) VsT[buf][c0 + j][r]     = vv0[j];
#pragma unroll
    for (int j = 0; j < 8; ++j) VsT[buf][c0 + 8 + j][r] = vv1[j];
  };

  const int nkb = (q0 + 128) / KB;   // q0/64 + 2
  stageKV(0, 0);
  for (int kbI = 0; kbI < nkb; ++kbI) {
    const int kbeg = kbI * KB;
    wait_asynccnt0();
    __syncthreads();                 // publish stage(kbI)
    if (kbI + 1 < nkb) stageKV((kbI + 1) & 1, (kbI + 1) * KB);
    if (kbeg > qt + 15) continue;    // wave-uniform; barriers all executed
    const int buf = kbI & 1;

    // scores: 16x64 = four 16x16 tiles, K-dim = HD via 2 WMMA each
    v8f s[4];
#pragma unroll
    for (int nt = 0; nt < 4; ++nt) {
      s[nt] = zero8();
#pragma unroll
      for (int kf = 0; kf < 2; ++kf) {
        const bf16* cp = &Ks[buf][nt * 16 + lr][kf * 32 + hf * 8];
        v16bf kfr = cat8(*(const v8bf*)cp, *(const v8bf*)(cp + 16));
        s[nt] = wmma_bf16(qa[kf], kfr, s[nt]);
      }
    }

    // scale + causal mask (row = qt + hf*8 + v, col = kbeg + nt*16 + lr)
#pragma unroll
    for (int nt = 0; nt < 4; ++nt)
#pragma unroll
      for (int v = 0; v < 8; ++v) {
        const float sv = s[nt][v] * scale;
        s[nt][v] = (kbeg + nt * 16 + lr > qt + hf * 8 + v) ? -1e30f : sv;
      }

    // row max: combine tiles per lane, then 4 shuffle steps, 8 rows in flight
    float mx[8];
#pragma unroll
    for (int v = 0; v < 8; ++v)
      mx[v] = fmaxf(fmaxf(s[0][v], s[1][v]), fmaxf(s[2][v], s[3][v]));
#pragma unroll
    for (int off = 1; off < 16; off <<= 1) {
#pragma unroll
      for (int v = 0; v < 8; ++v)
        mx[v] = fmaxf(mx[v], __shfl_xor(mx[v], off, 16));
    }

    // online update: rescale, exponentiate, spill P to LDS (bf16)
#pragma unroll
    for (int v = 0; v < 8; ++v) {
      const float nm    = fmaxf(rmax[v], mx[v]);
      const float alpha = __expf(rmax[v] - nm);
      rmax[v] = nm;
      rsum[v] *= alpha;
#pragma unroll
      for (int nt = 0; nt < 4; ++nt) {
        const float p = __expf(s[nt][v] - nm);
        Ps[wave][hf * 8 + v][nt * 16 + lr] = tobf(p);
      }
#pragma unroll
      for (int nt = 0; nt < 4; ++nt) o[nt][v] *= alpha;
    }

    // reload P as A-fragments (same-wave LDS is in order)
    const bf16* prow = &Ps[wave][lr][0];
    v16bf pa0 = cat8(*(const v8bf*)(prow + hf * 8),
                     *(const v8bf*)(prow + 16 + hf * 8));
    v16bf pa1 = cat8(*(const v8bf*)(prow + 32 + hf * 8),
                     *(const v8bf*)(prow + 48 + hf * 8));

    // row sums via WMMA against all-ones B: D[m][*] = sum_k P[m][k]
    v8f psum = wmma_bf16(pa0, ones, zero8());
    psum     = wmma_bf16(pa1, ones, psum);
#pragma unroll
    for (int v = 0; v < 8; ++v) rsum[v] += psum[v];

    // P (16x64) @ V (64x64)
#pragma unroll
    for (int nt = 0; nt < 4; ++nt) {
      const bf16* vp0 = &VsT[buf][nt * 16 + lr][hf * 8];
      const bf16* vp1 = &VsT[buf][nt * 16 + lr][32 + hf * 8];
      v16bf vf0 = cat8(*(const v8bf*)vp0, *(const v8bf*)(vp0 + 16));
      v16bf vf1 = cat8(*(const v8bf*)vp1, *(const v8bf*)(vp1 + 16));
      o[nt] = wmma_bf16(pa0, vf0, o[nt]);
      o[nt] = wmma_bf16(pa1, vf1, o[nt]);
    }
  }

  // normalize, store bf16
#pragma unroll
  for (int v = 0; v < 8; ++v) {
    const int qrow = qt + hf * 8 + v;
    const float inv = 1.0f / rsum[v];
    bf16* orow = ob + (long)qrow * H_;
#pragma unroll
    for (int nt = 0; nt < 4; ++nt) orow[nt * 16 + lr] = tobf(o[nt][v] * inv);
  }
}

// ---------------------------------------------------------------------------
extern "C" void kernel_launch(void* const* d_in, const int* in_sizes, int n_in,
                              void* d_out, int out_size, void* d_ws, size_t ws_size,
                              hipStream_t stream) {
  (void)in_sizes; (void)n_in; (void)out_size; (void)ws_size;
  const float* X  = (const float*)d_in[0];
  // d_in[1] = attention_mask: causal, known statically -> ignored
  const float* Wq = (const float*)d_in[2];
  const float* bq = (const float*)d_in[3];
  const float* Wk = (const float*)d_in[4];
  const float* bk = (const float*)d_in[5];
  const float* Wv = (const float*)d_in[6];
  const float* bv = (const float*)d_in[7];
  const float* Wo = (const float*)d_in[8];
  const float* bo = (const float*)d_in[9];
  float* out = (float*)d_out;

  // bf16 intermediates in workspace
  bf16* Qb = (bf16*)d_ws;                 // 4096*1024
  bf16* Kb = Qb + (size_t)4096 * 1024;    // 4096*256
  bf16* Vb = Kb + (size_t)4096 * 256;     // 4096*256
  bf16* Ab = Vb + (size_t)4096 * 256;     // 4096*1024

  const int M = B_ * S_;   // 4096
  dim3 blk(256);

  gemm_bias_kernel<float, true><<<dim3(M / 128, H_ / 128),   blk, 0, stream>>>(X,  Wq, bq, Qb,  M, H_,   H_);
  gemm_bias_kernel<float, true><<<dim3(M / 128, KVW_ / 128), blk, 0, stream>>>(X,  Wk, bk, Kb,  M, KVW_, H_);
  gemm_bias_kernel<float, true><<<dim3(M / 128, KVW_ / 128), blk, 0, stream>>>(X,  Wv, bv, Vb,  M, KVW_, H_);
  gqa_flash_kernel<<<dim3(S_ / 128, NH_, B_), blk, 0, stream>>>(Qb, Kb, Vb, Ab);
  gemm_bias_kernel<bf16, false><<<dim3(M / 128, H_ / 128),   blk, 0, stream>>>(Ab, Wo, bo, out, M, H_,   H_);
}
